// Block_ViT_1554778161854
// MI455X (gfx1250) — matrix operand order repaired and checked by reading, not compiled, not run on hardware
//
#include <hip/hip_runtime.h>
#include <hip/hip_bf16.h>
#include <math.h>

typedef __attribute__((ext_vector_type(16))) _Float16 v16h;
typedef __attribute__((ext_vector_type(8)))  _Float16 v8h;
typedef __attribute__((ext_vector_type(8)))  float    v8f;
typedef __attribute__((ext_vector_type(4)))  float    v4f;

// ---------------------------------------------------------------------------
// Tiled WMMA GEMM:  C[b] = alpha * A[b] (MxK) * op(B[b]) (+ Res[b])
//   BT == false : B is (K x N) row-major (ldb = row stride)
//   BT == true  : B is (N x K) row-major (inner product over K)  => C = A*B^T
// 256 threads = 8 waves; block tile 32x128. Each wave owns two 16x16 C tiles
// (cols n and n+64) sharing one A fragment -> 2 v_wmma per K-step.
// Tiles staged in LDS as f16 (converted from f32). A stored [m][k]; B stored
// transposed [n][k] so every fragment is two contiguous 16B runs
// (2x ds_load_b128 per fragment). Requires M % 32 == 0 and K % 32 == 0.
// ---------------------------------------------------------------------------
constexpr int TM = 32, TN = 128, TK = 32;

#define CAT16(lo, hi) __builtin_shufflevector(lo, hi, 0,1,2,3,4,5,6,7,8,9,10,11,12,13,14,15)

template<bool BT>
__global__ void __launch_bounds__(256)
gemm_wmma(const float* __restrict__ A, const float* __restrict__ Bm,
          const float* __restrict__ Res, float* __restrict__ Cm,
          int M, int Ncols, int K, int lda, int ldb, int ldc,
          size_t sA, size_t sB, size_t sC, float alpha)
{
    // row pitch = TK+8 = 40 halves = 80 bytes (multiple of 16 -> b128-aligned)
    __shared__ __align__(16) _Float16 As[TM][TK + 8];   // [m][k]
    __shared__ __align__(16) _Float16 Bs[TN][TK + 8];   // [n][k] (transposed)

    const int tid  = threadIdx.x;
    const int lane = tid & 31;
    const int wave = tid >> 5;
    const int tm   = (wave >> 2) * 16;    // 0 or 16
    const int tn   = (wave & 3) * 16;     // 0,16,32,48 (second tile at +64)
    const int g    = lane >> 4;           // lane group
    const int mr   = lane & 15;

    const int m0 = blockIdx.y * TM;
    const int n0 = blockIdx.x * TN;
    const int bz = blockIdx.z;

    const float* Ab = A  + (size_t)bz * sA;
    const float* Bb = Bm + (size_t)bz * sB;

    v8f acc0 = {}, acc1 = {};

    for (int kk = 0; kk < K; kk += TK) {
        // ---- stage A tile (TM x TK): float4 load + 4 cvt + b64 store ----
        {
            int r  = tid >> 3;            // 0..31
            int c4 = (tid & 7) * 4;       // 0..28
            v4f v = *(const v4f*)(Ab + (size_t)(m0 + r) * lda + kk + c4);
            #pragma unroll
            for (int u = 0; u < 4; ++u)
                As[r][c4 + u] = (_Float16)v[u];
        }
        // ---- stage B tile (TK x TN) into Bs[n][k] ----
        if (BT) {
            int n   = tid >> 1;           // 0..127
            int k16 = (tid & 1) * 16;     // 0 or 16
            int ncl = n0 + n;
            bool ok = ncl < Ncols;
            if (!ok) ncl = Ncols - 1;     // clamped (legal) address; zero value
            const float* Br = Bb + (size_t)ncl * ldb + kk + k16;
            #pragma unroll
            for (int q = 0; q < 4; ++q) {
                v4f v = *(const v4f*)(Br + 4 * q);
                #pragma unroll
                for (int u = 0; u < 4; ++u)
                    Bs[n][k16 + 4 * q + u] = (_Float16)(ok ? v[u] : 0.f);
            }
        } else {
            int n   = tid & 127;          // 0..127
            int k16 = (tid >> 7) * 16;    // 0 or 16
            int ncl = n0 + n;
            bool ok = ncl < Ncols;
            if (!ok) ncl = Ncols - 1;
            #pragma unroll
            for (int u = 0; u < 16; ++u) {
                float v = Bb[(size_t)(kk + k16 + u) * ldb + ncl];
                Bs[n][k16 + u] = (_Float16)(ok ? v : 0.f);
            }
        }
        __syncthreads();

        // ---- fragments (contiguous k = 16g+j order, same in A and B), 2x b128 each ----
        const v8h* arow = (const v8h*)&As[tm + mr][0];          // 5 v8h / row
    v16h af = CAT16(arow[2 * g], arow[2 * g + 1]);          // K = 16g+j

        const v8h* br0 = (const v8h*)&Bs[tn + mr][0];
        v16h bf0 = CAT16(br0[2 * g], br0[2 * g + 1]);           // K = 16g+j
        const v8h* br1 = (const v8h*)&Bs[64 + tn + mr][0];
        v16h bf1 = CAT16(br1[2 * g], br1[2 * g + 1]);

        acc0 = __builtin_amdgcn_wmma_f32_16x16x32_f16(
                   false, af, false, bf0, (short)0, acc0, false, false);
        acc1 = __builtin_amdgcn_wmma_f32_16x16x32_f16(
                   false, af, false, bf1, (short)0, acc1, false, false);
        __syncthreads();
    }

    // ---- epilogue: C/D layout -> lane n = l&15, VGPR r -> m = r + 8*g ----
    {
        int n = n0 + tn + mr;
        if (n < Ncols) {
            #pragma unroll
            for (int r = 0; r < 8; ++r) {
                int m = m0 + tm + r + 8 * g;
                float v = acc0[r] * alpha;
                if (Res) v += Res[(size_t)bz * sC + (size_t)m * ldc + n];
                Cm[(size_t)bz * sC + (size_t)m * ldc + n] = v;
            }
        }
    }
    {
        int n = n0 + 64 + tn + mr;
        if (n < Ncols) {
            #pragma unroll
            for (int r = 0; r < 8; ++r) {
                int m = m0 + tm + r + 8 * g;
                float v = acc1[r] * alpha;
                if (Res) v += Res[(size_t)bz * sC + (size_t)m * ldc + n];
                Cm[(size_t)bz * sC + (size_t)m * ldc + n] = v;
            }
        }
    }
}

// ---------------------------------------------------------------------------
// Elementwise / reduction kernels (bandwidth-bound paths)
// ---------------------------------------------------------------------------
__global__ void __launch_bounds__(256)
concat_copy(const float* __restrict__ src, float* __restrict__ dst,
            int C, int choff, int Np, int KVt, size_t total)
{
    size_t i = (size_t)blockIdx.x * 256 + threadIdx.x;
    if (i >= total) return;
    size_t per = (size_t)C * Np;
    size_t b = i / per, r = i - b * per;
    dst[b * (size_t)KVt * Np + (size_t)choff * Np + r] = src[i];
}

// LayerNorm over channel dim per (b, pixel); channel stride == Np.
__global__ void __launch_bounds__(256)
ln_kernel(const float* __restrict__ x, const float* __restrict__ w,
          const float* __restrict__ bia, float* __restrict__ y,
          int C, int Np, size_t sX, size_t sY)
{
    int b = blockIdx.y;
    int n = blockIdx.x * 256 + threadIdx.x;
    if (n >= Np) return;
    const float* xb = x + (size_t)b * sX + n;
    float mu = 0.f;
    for (int c = 0; c < C; ++c) mu += xb[(size_t)c * Np];
    mu /= (float)C;
    float var = 0.f;
    for (int c = 0; c < C; ++c) { float d = xb[(size_t)c * Np] - mu; var += d * d; }
    var /= (float)C;
    float inv = rsqrtf(var + 1e-5f);
    float* yb = y + (size_t)b * sY + n;
    for (int c = 0; c < C; ++c)
        yb[(size_t)c * Np] = (xb[(size_t)c * Np] - mu) * inv * w[c] + bia[c];
}

// Grouped conv, gin input channels per group (gin==out-per-group here: 1 or 2)
__global__ void __launch_bounds__(256)
gconv_kernel(const float* __restrict__ x, const float* __restrict__ w,
             float* __restrict__ y, int Cout, int gin, int ks, int pad, int relu,
             size_t sX, size_t sY, int H, int W)
{
    int b = blockIdx.z, o = blockIdx.y;
    int n = blockIdx.x * 256 + threadIdx.x;
    if (n >= H * W) return;
    int py = n / W, px = n % W;
    int ib = (o / gin) * gin;
    float acc = 0.f;
    for (int ic = 0; ic < gin; ++ic) {
        const float* xi = x + (size_t)b * sX + (size_t)(ib + ic) * H * W;
        const float* wk = w + ((size_t)o * gin + ic) * ks * ks;
        for (int ky = 0; ky < ks; ++ky) {
            int yy = py + ky - pad;
            if (yy < 0 || yy >= H) continue;
            for (int kx = 0; kx < ks; ++kx) {
                int xx = px + kx - pad;
                if (xx < 0 || xx >= W) continue;
                acc += xi[yy * W + xx] * wk[ky * ks + kx];
            }
        }
    }
    if (relu) acc = fmaxf(acc, 0.f);
    y[(size_t)b * sY + (size_t)o * H * W + n] = acc;
}

// In-place L2 normalize each (b, c) row of length Np.
__global__ void __launch_bounds__(256)
l2norm_rows(float* __restrict__ x, int Np, size_t sB)
{
    int b = blockIdx.y, c = blockIdx.x;
    float* row = x + (size_t)b * sB + (size_t)c * Np;
    float s = 0.f;
    for (int i = threadIdx.x; i < Np; i += 256) { float v = row[i]; s += v * v; }
    __shared__ float red[256];
    red[threadIdx.x] = s; __syncthreads();
    for (int o = 128; o > 0; o >>= 1) {
        if (threadIdx.x < o) red[threadIdx.x] += red[threadIdx.x + o];
        __syncthreads();
    }
    float inv = 1.f / fmaxf(sqrtf(red[0]), 1e-12f);
    for (int i = threadIdx.x; i < Np; i += 256) row[i] *= inv;
}

__global__ void __launch_bounds__(256)
attn_stats(const float* __restrict__ a, float* __restrict__ st, int count)
{
    int b = blockIdx.x;
    const float* ab = a + (size_t)b * count;
    float s = 0.f, s2 = 0.f;
    for (int i = threadIdx.x; i < count; i += 256) { float v = ab[i]; s += v; s2 += v * v; }
    __shared__ float r1[256], r2[256];
    r1[threadIdx.x] = s; r2[threadIdx.x] = s2; __syncthreads();
    for (int o = 128; o > 0; o >>= 1) {
        if (threadIdx.x < o) { r1[threadIdx.x] += r1[threadIdx.x + o];
                               r2[threadIdx.x] += r2[threadIdx.x + o]; }
        __syncthreads();
    }
    if (threadIdx.x == 0) {
        float mu = r1[0] / (float)count;
        st[b * 2]     = mu;
        st[b * 2 + 1] = r2[0] / (float)count - mu * mu;
    }
}

// In-place standardized softmax over k (row length KVn).
__global__ void __launch_bounds__(256)
attn_softmax(float* __restrict__ a, const float* __restrict__ st, int C, int KVn)
{
    int b = blockIdx.y, c = blockIdx.x;
    float* row = a + ((size_t)b * C + c) * KVn;
    float mu  = st[b * 2];
    float inv = rsqrtf(st[b * 2 + 1] + 1e-5f);
    __shared__ float red[256];
    float mx = -1e30f;
    for (int k = threadIdx.x; k < KVn; k += 256)
        mx = fmaxf(mx, (row[k] - mu) * inv);
    red[threadIdx.x] = mx; __syncthreads();
    for (int o = 128; o > 0; o >>= 1) {
        if (threadIdx.x < o) red[threadIdx.x] = fmaxf(red[threadIdx.x], red[threadIdx.x + o]);
        __syncthreads();
    }
    mx = red[0]; __syncthreads();
    float sum = 0.f;
    for (int k = threadIdx.x; k < KVn; k += 256)
        sum += __expf((row[k] - mu) * inv - mx);
    red[threadIdx.x] = sum; __syncthreads();
    for (int o = 128; o > 0; o >>= 1) {
        if (threadIdx.x < o) red[threadIdx.x] += red[threadIdx.x + o];
        __syncthreads();
    }
    float rs = 1.f / red[0];
    for (int k = threadIdx.x; k < KVn; k += 256)
        row[k] = __expf((row[k] - mu) * inv - mx) * rs;
}

__global__ void __launch_bounds__(256)
rowmean(const float* __restrict__ x, float* __restrict__ s, int Np, size_t sB, int C)
{
    int b = blockIdx.y, c = blockIdx.x;
    const float* row = x + (size_t)b * sB + (size_t)c * Np;
    float v = 0.f;
    for (int i = threadIdx.x; i < Np; i += 256) v += row[i];
    __shared__ float red[256];
    red[threadIdx.x] = v; __syncthreads();
    for (int o = 128; o > 0; o >>= 1) {
        if (threadIdx.x < o) red[threadIdx.x] += red[threadIdx.x + o];
        __syncthreads();
    }
    if (threadIdx.x == 0) s[b * C + c] = red[0] / (float)Np;
}

// out = y * sigmoid(eca(s)) + resid
__global__ void __launch_bounds__(256)
eca_apply(const float* __restrict__ y, const float* __restrict__ s,
          const float* __restrict__ we, const float* __restrict__ resid,
          float* __restrict__ out, int C, int Np, size_t sY, size_t sR, size_t sO)
{
    int b = blockIdx.z, c = blockIdx.y;
    int n = blockIdx.x * 256 + threadIdx.x;
    if (n >= Np) return;
    float sm1 = (c > 0)     ? s[b * C + c - 1] : 0.f;
    float s0  =               s[b * C + c];
    float sp1 = (c < C - 1) ? s[b * C + c + 1] : 0.f;
    float z = we[0] * sm1 + we[1] * s0 + we[2] * sp1;
    float gv = 1.f / (1.f + __expf(-z));
    out[(size_t)b * sO + (size_t)c * Np + n] =
        y[(size_t)b * sY + (size_t)c * Np + n] * gv +
        resid[(size_t)b * sR + (size_t)c * Np + n];
}

// ---------------------------------------------------------------------------
// Orchestration
// ---------------------------------------------------------------------------
extern "C" void kernel_launch(void* const* d_in, const int* in_sizes, int n_in,
                              void* d_out, int out_size, void* d_ws, size_t ws_size,
                              hipStream_t stream)
{
    (void)in_sizes; (void)n_in; (void)out_size; (void)ws_size;

    constexpr int Bn = 4, Hh = 64, Ww = 64, Np = 64 * 64, KVc = 480;
    constexpr int CHs[4] = {32, 64, 128, 256};
    constexpr int CO[4]  = {0, 32, 96, 224};
    const size_t SB = (size_t)KVc * Np;          // per-batch stride of concat buffers
    const size_t NB = (size_t)Bn * SB;           // full concat buffer

    // Input index map (setup_inputs flatten order):
    // 0..3: emb1..emb4; per scale i (stride 12 from 4):
    //   +0 ln_attn_w +1 ln_attn_b +2 ln_ffn_w +3 ln_ffn_b +4 mhead +5 q
    //   +6 proj +7 w_in +8 w3 +9 w5 +10 w_out +11 w_eca
    // 52 ln_attnK_w, 53 ln_attnK_b, 54 mheadk, 55 mheadv, 56 k, 57 v
    auto P = [&](int idx) { return (const float*)d_in[idx]; };
    const float* emb[4] = { P(0), P(1), P(2), P(3) };

    float* ws = (float*)d_ws;
    float* f_emb  = ws;                                   // concat emb (residual)
    float* f_ea   = ws + 1 * NB;                          // LN_K(concat)
    float* f_cx   = ws + 2 * NB;                          // per-scale LN
    float* f_q    = ws + 3 * NB;
    float* f_k    = ws + 4 * NB;
    float* f_v    = ws + 5 * NB;
    float* f_o    = ws + 6 * NB;
    float* f_out  = ws + 7 * NB;                          // attn branch out (+resid)
    float* f_t1   = ws + 8 * NB;                          // B*2048*Np region
    float* f_y    = f_t1 + (size_t)Bn * 2048 * Np;        // B*256*Np
    float* f_ln   = f_y  + (size_t)Bn * 256 * Np;         // B*256*Np
    float* f_attn = f_ln + (size_t)Bn * 256 * Np;         // B*256*KVc
    float* f_st   = f_attn + (size_t)Bn * 256 * KVc;      // stats (8 fl)
    float* f_s    = f_st + 16;                            // ECA means (<=1024 fl)
    float* f_tA   = f_t1;                                 // attn-phase 1x1 tmp (1*NB)
    float* f_t0   = ws;                                   // FFN w_in out: reuses dead bufs

    dim3 b256(256);
    dim3 gpix(Np / 256, Bn);

    auto gemm = [&](bool bt, const float* A, const float* Bm, const float* Res,
                    float* C, int M, int N, int K, int lda, int ldb, int ldc,
                    size_t sA, size_t sBm, size_t sC, float alpha) {
        dim3 g((N + TN - 1) / TN, M / TM, Bn);
        if (bt) gemm_wmma<true ><<<g, b256, 0, stream>>>(A, Bm, Res, C, M, N, K,
                                                         lda, ldb, ldc, sA, sBm, sC, alpha);
        else    gemm_wmma<false><<<g, b256, 0, stream>>>(A, Bm, Res, C, M, N, K,
                                                         lda, ldb, ldc, sA, sBm, sC, alpha);
    };

    // 1) concat embeddings
    for (int i = 0; i < 4; ++i) {
        size_t tot = (size_t)Bn * CHs[i] * Np;
        concat_copy<<<dim3((unsigned)((tot + 255) / 256)), b256, 0, stream>>>(
            emb[i], f_emb, CHs[i], CO[i], Np, KVc, tot);
    }

    // 2) LayerNorms
    for (int i = 0; i < 4; ++i)
        ln_kernel<<<gpix, b256, 0, stream>>>(f_emb + (size_t)CO[i] * Np,
            P(4 + 12 * i + 0), P(4 + 12 * i + 1),
            f_cx + (size_t)CO[i] * Np, CHs[i], Np, SB, SB);
    ln_kernel<<<gpix, b256, 0, stream>>>(f_emb, P(52), P(53), f_ea, KVc, Np, SB, SB);

    // 3) q = gconv3(group=2)(mhead @ cx)
    for (int i = 0; i < 4; ++i) {
        int c = CHs[i];
        gemm(false, P(4 + 12 * i + 4), f_cx + (size_t)CO[i] * Np, nullptr,
             f_tA + (size_t)CO[i] * Np, c, Np, c, c, Np, Np, 0, SB, SB, 1.f);
        gconv_kernel<<<dim3(Np / 256, c, Bn), b256, 0, stream>>>(
            f_tA + (size_t)CO[i] * Np, P(4 + 12 * i + 5), f_q + (size_t)CO[i] * Np,
            c, 2, 3, 1, 0, SB, SB, Hh, Ww);
    }

    // 4) k = dw3(mheadk @ ea), v = dw3(mheadv @ ea)
    gemm(false, P(54), f_ea, nullptr, f_tA, KVc, Np, KVc, KVc, Np, Np, 0, SB, SB, 1.f);
    gconv_kernel<<<dim3(Np / 256, KVc, Bn), b256, 0, stream>>>(
        f_tA, P(56), f_k, KVc, 1, 3, 1, 0, SB, SB, Hh, Ww);
    gemm(false, P(55), f_ea, nullptr, f_tA, KVc, Np, KVc, KVc, Np, Np, 0, SB, SB, 1.f);
    gconv_kernel<<<dim3(Np / 256, KVc, Bn), b256, 0, stream>>>(
        f_tA, P(57), f_v, KVc, 1, 3, 1, 0, SB, SB, Hh, Ww);

    // 5) l2norm over spatial for kn and qn
    l2norm_rows<<<dim3(KVc, Bn), b256, 0, stream>>>(f_k, Np, SB);
    l2norm_rows<<<dim3(KVc, Bn), b256, 0, stream>>>(f_q, Np, SB);

    // 6) attention per scale:  attn = (Qn Kn^T)/sqrt(KV); standardize; softmax;
    //    o = P V; out = proj(o) + emb
    const float alpha = 1.f / sqrtf((float)KVc);
    for (int i = 0; i < 4; ++i) {
        int c = CHs[i];
        gemm(true, f_q + (size_t)CO[i] * Np, f_k, nullptr, f_attn,
             c, KVc, Np, Np, Np, KVc, SB, SB, (size_t)c * KVc, alpha);
        attn_stats<<<dim3(Bn), b256, 0, stream>>>(f_attn, f_st, c * KVc);
        attn_softmax<<<dim3(c, Bn), b256, 0, stream>>>(f_attn, f_st, c, KVc);
        gemm(false, f_attn, f_v, nullptr, f_o + (size_t)CO[i] * Np,
             c, Np, KVc, KVc, Np, Np, (size_t)c * KVc, SB, SB, 1.f);
        gemm(false, P(4 + 12 * i + 6), f_o + (size_t)CO[i] * Np,
             f_emb + (size_t)CO[i] * Np, f_out + (size_t)CO[i] * Np,
             c, Np, c, c, Np, Np, 0, SB, SB, 1.f);
    }

    // 7) FFN per scale (reuses attention-phase buffers for big temporaries)
    for (int i = 0; i < 4; ++i) {
        int c = CHs[i], hid = 4 * c, h2 = 8 * c;
        size_t sLn = (size_t)c * Np, sT = (size_t)h2 * Np;
        ln_kernel<<<gpix, b256, 0, stream>>>(f_out + (size_t)CO[i] * Np,
            P(4 + 12 * i + 2), P(4 + 12 * i + 3), f_ln, c, Np, SB, sLn);
        gemm(false, P(4 + 12 * i + 7), f_ln, nullptr, f_t0,
             h2, Np, c, c, Np, Np, 0, sLn, sT, 1.f);
        gconv_kernel<<<dim3(Np / 256, hid, Bn), b256, 0, stream>>>(
            f_t0, P(4 + 12 * i + 8), f_t1, hid, 1, 3, 1, 1, sT, sT, Hh, Ww);
        gconv_kernel<<<dim3(Np / 256, hid, Bn), b256, 0, stream>>>(
            f_t0 + (size_t)hid * Np, P(4 + 12 * i + 9), f_t1 + (size_t)hid * Np,
            hid, 1, 5, 2, 1, sT, sT, Hh, Ww);
        gemm(false, P(4 + 12 * i + 10), f_t1, nullptr, f_y,
             c, Np, h2, h2, Np, Np, 0, sT, sLn, 1.f);
        rowmean<<<dim3(c, Bn), b256, 0, stream>>>(f_y, f_s, Np, sLn, c);
        float* outp = (float*)d_out + (size_t)Bn * Np * CO[i];
        eca_apply<<<dim3(Np / 256, c, Bn), b256, 0, stream>>>(
            f_y, f_s, P(4 + 12 * i + 11), f_out + (size_t)CO[i] * Np, outp,
            c, Np, sLn, SB, sLn);
    }
}
